// VectorQuantiser_56075093017340
// MI455X (gfx1250) — matrix operand order repaired and checked by reading, not compiled, not run on hardware
//
#include <hip/hip_runtime.h>
#include <hip/hip_bf16.h>

typedef __attribute__((ext_vector_type(16))) __bf16 v16bf;
typedef __attribute__((ext_vector_type(8)))  float  v8f;
typedef __attribute__((ext_vector_type(4)))  unsigned int v4u;
typedef __attribute__((ext_vector_type(8)))  int v8i;
typedef __attribute__((ext_vector_type(4)))  int v4i;

#define B_ 8
#define T_ 4096
#define D_ 512
#define K_ 4096
#define N_ (B_ * T_)   // 32768 rows

#define TILE_ROWS 128
#define TILE_COLS 32   // D-chunk depth (one K=32 WMMA step)

#if __has_include(<hip/amd_detail/amd_gfx1250_TDM.h>)
#define TDM_6ARG 1
#else
#define TDM_6ARG 0
#endif

// ---- f32 <-> bf16 split helpers (round-to-nearest-even) ----
__device__ __forceinline__ unsigned short f2bf_rne(float x) {
    unsigned int b = __float_as_uint(x);
    b += 0x7FFFu + ((b >> 16) & 1u);
    return (unsigned short)(b >> 16);
}
__device__ __forceinline__ float bf2f(unsigned short h) {
    return __uint_as_float(((unsigned int)h) << 16);
}

// ---- TDM: 2D tile load (TILE_ROWS x TILE_COLS bf16) global -> LDS ----
// D# per CDNA5 ISA: group0 {count=1, lds_addr,
// global_addr, type=2}; group1 {data_size=2B, tensor_dim0/1, tile_dim0/1,
// tensor_dim0_stride}; groups 2/3 zero (2D tensor).
__device__ __forceinline__ void tdm_load_tile(unsigned lds_byte_off,
                                              const unsigned short* gsrc,
                                              unsigned row_stride_elems) {
    unsigned long long ga = (unsigned long long)(uintptr_t)gsrc;
    v4u g0;
    g0[0] = 1u;                                    // count=1, user descriptor
    g0[1] = lds_byte_off;                          // lds_addr (bytes)
    g0[2] = (unsigned)(ga & 0xFFFFFFFFull);        // global_addr[31:0]
    g0[3] = (unsigned)((ga >> 32) & 0x1FFFFFFu) | (2u << 30);  // addr[56:32], type=2
    v8i g1;
    g1[0] = (1 << 16);                             // workgroup_mask=0, data_size=1 (2B)
    g1[1] = (int)(((unsigned)TILE_COLS & 0xFFFFu) << 16);          // tensor_dim0 lo16
    g1[2] = (int)((((unsigned)TILE_COLS >> 16) & 0xFFFFu)
                  | (((unsigned)TILE_ROWS & 0xFFFFu) << 16));      // dim0 hi / dim1 lo
    g1[3] = (int)((((unsigned)TILE_ROWS >> 16) & 0xFFFFu)
                  | ((unsigned)TILE_COLS << 16));                  // dim1 hi / tile_dim0
    g1[4] = (int)((unsigned)TILE_ROWS & 0xFFFFu);                  // tile_dim1; tile_dim2=0
    g1[5] = (int)row_stride_elems;                                 // tensor_dim0_stride lo32
    g1[6] = 0;                                                     // stride hi / dim1_stride lo
    g1[7] = 0;
    v4i gz = {0, 0, 0, 0};
#if TDM_6ARG
    v8i gz8 = {0, 0, 0, 0, 0, 0, 0, 0};
    __builtin_amdgcn_tensor_load_to_lds(g0, g1, gz, gz, gz8, 0);
#else
    __builtin_amdgcn_tensor_load_to_lds(g0, g1, gz, gz, 0);
#endif
}

// ============================================================
// Kernel 0: split f32 -> (bf16 hi, bf16 lo) planes, 4 elems/thread.
// ============================================================
__global__ void vq_split(const float* __restrict__ src,
                         unsigned short* __restrict__ hi,
                         unsigned short* __restrict__ lo) {
    size_t i = ((size_t)blockIdx.x * 256 + threadIdx.x) * 4;
    float4 v = *reinterpret_cast<const float4*>(src + i);
    float f[4] = {v.x, v.y, v.z, v.w};
    unsigned int hh[4], ll[4];
    #pragma unroll
    for (int q = 0; q < 4; ++q) {
        unsigned short a = f2bf_rne(f[q]);
        hh[q] = a;
        ll[q] = f2bf_rne(f[q] - bf2f(a));
    }
    uint2 hs, ls;
    hs.x = hh[0] | (hh[1] << 16);  hs.y = hh[2] | (hh[3] << 16);
    ls.x = ll[0] | (ll[1] << 16);  ls.y = ll[2] | (ll[3] << 16);
    *reinterpret_cast<uint2*>(hi + i) = hs;
    *reinterpret_cast<uint2*>(lo + i) = ls;
}

// ============================================================
// Kernel 1: c2[k] = sum(codebook[k,:]^2); zero loss accumulator.
// ============================================================
__global__ void vq_prep(const float* __restrict__ cb,
                        float* __restrict__ c2,
                        float* __restrict__ wloss) {
    const int wave = threadIdx.x >> 5;
    const int lane = threadIdx.x & 31;
    const int k = blockIdx.x * 8 + wave;
    const float* row = cb + (size_t)k * D_;
    float s = 0.f;
    #pragma unroll
    for (int j = 0; j < 16; ++j) {
        float v = row[lane + j * 32];
        s = __builtin_fmaf(v, v, s);
    }
    #pragma unroll
    for (int m = 16; m >= 1; m >>= 1) s += __shfl_xor(s, m, 32);
    if (lane == 0) c2[k] = s;
    if (blockIdx.x == 0 && threadIdx.x == 0) *wloss = 0.f;
}

// ============================================================
// Kernel 2: fused GEMM + argmin, TDM-staged, double-buffered.
// Block: 256 threads (8 waves), owns 128 rows. Loops 128-col
// tiles; D in 32-deep chunks (one K=32 step per chunk).
// Wave 0 drives the Tensor Data Mover; TENSORcnt fences buffers.
// ============================================================
__launch_bounds__(256, 2)
__global__ void vq_argmin(const unsigned short* __restrict__ h_hi,
                          const unsigned short* __restrict__ h_lo,
                          const unsigned short* __restrict__ cb_hi,
                          const unsigned short* __restrict__ cb_lo,
                          const float* __restrict__ c2,
                          int* __restrict__ outidx) {
    // 2 buffers x 4 tiles x (128x32) bf16 = 64 KB total.
    __shared__ __align__(16) unsigned short As_hi[2][TILE_ROWS][TILE_COLS];
    __shared__ __align__(16) unsigned short As_lo[2][TILE_ROWS][TILE_COLS];
    __shared__ __align__(16) unsigned short Bs_hi[2][TILE_ROWS][TILE_COLS];
    __shared__ __align__(16) unsigned short Bs_lo[2][TILE_ROWS][TILE_COLS];

    const int tid     = threadIdx.x;
    const int wave    = tid >> 5;        // 0..7 -> 16-row strip
    const int lane    = tid & 31;
    const int halfsel = lane >> 4;
    const int ln      = lane & 15;
    const int rowbase = blockIdx.x * 128;
    const bool issuer = (wave == 0);

    float minv[8];
    int   mini[8];
    #pragma unroll
    for (int r = 0; r < 8; ++r) { minv[r] = 3.4e38f; mini[r] = 0; }

    for (int ct = 0; ct < K_; ct += 128) {
        v8f acc[8];
        #pragma unroll
        for (int t = 0; t < 8; ++t)
            #pragma unroll
            for (int r = 0; r < 8; ++r) acc[t][r] = 0.f;

        // prime buffer 0 with chunk 0
        if (issuer) {
            const size_t aoffg = (size_t)rowbase * D_;
            const size_t boffg = (size_t)ct * D_;
            tdm_load_tile((unsigned)(uintptr_t)&As_hi[0][0][0], h_hi + aoffg, D_);
            tdm_load_tile((unsigned)(uintptr_t)&As_lo[0][0][0], h_lo + aoffg, D_);
            tdm_load_tile((unsigned)(uintptr_t)&Bs_hi[0][0][0], cb_hi + boffg, D_);
            tdm_load_tile((unsigned)(uintptr_t)&Bs_lo[0][0][0], cb_lo + boffg, D_);
        }

        for (int chunk = 0; chunk < D_ / TILE_COLS; ++chunk) {
            const int buf = chunk & 1;
            if (issuer) {
                if (chunk < D_ / TILE_COLS - 1) {
                    const int nb = buf ^ 1;
                    const size_t dnext = (size_t)(chunk + 1) * TILE_COLS;
                    const size_t aoffg = (size_t)rowbase * D_ + dnext;
                    const size_t boffg = (size_t)ct * D_ + dnext;
                    tdm_load_tile((unsigned)(uintptr_t)&As_hi[nb][0][0], h_hi + aoffg, D_);
                    tdm_load_tile((unsigned)(uintptr_t)&As_lo[nb][0][0], h_lo + aoffg, D_);
                    tdm_load_tile((unsigned)(uintptr_t)&Bs_hi[nb][0][0], cb_hi + boffg, D_);
                    tdm_load_tile((unsigned)(uintptr_t)&Bs_lo[nb][0][0], cb_lo + boffg, D_);
                    __builtin_amdgcn_s_wait_tensorcnt(4);  // front buffer complete
                } else {
                    __builtin_amdgcn_s_wait_tensorcnt(0);  // all complete
                }
            }
            __syncthreads();

            // ---- WMMA on this 32-deep chunk ----
            // A fragment (16x32 bf16): lane ln = row M; 8-half chunks at
            // off and off+16, off = halfsel*8.
            const int arow = wave * 16 + ln;
            const int aoff = halfsel ? 8 : 0;
            union { uint4 u[2]; v16bf v; } fa_hi, fa_lo;
            fa_hi.u[0] = *reinterpret_cast<const uint4*>(&As_hi[buf][arow][aoff]);
            fa_hi.u[1] = *reinterpret_cast<const uint4*>(&As_hi[buf][arow][aoff + 16]);
            fa_lo.u[0] = *reinterpret_cast<const uint4*>(&As_lo[buf][arow][aoff]);
            fa_lo.u[1] = *reinterpret_cast<const uint4*>(&As_lo[buf][arow][aoff + 16]);

            #pragma unroll
            for (int t = 0; t < 8; ++t) {
                // B fragment (32x16 bf16): lane ln = column N (= codebook
                // row), 8-half chunks at off and off+16, off = halfsel*8.
                const int brow = t * 16 + ln;
                const int boff = halfsel ? 8 : 0;
                union { uint4 u[2]; v16bf v; } fb_hi, fb_lo;
                fb_hi.u[0] = *reinterpret_cast<const uint4*>(&Bs_hi[buf][brow][boff]);
                fb_hi.u[1] = *reinterpret_cast<const uint4*>(&Bs_hi[buf][brow][boff + 16]);
                fb_lo.u[0] = *reinterpret_cast<const uint4*>(&Bs_lo[buf][brow][boff]);
                fb_lo.u[1] = *reinterpret_cast<const uint4*>(&Bs_lo[buf][brow][boff + 16]);

                acc[t] = __builtin_amdgcn_wmma_f32_16x16x32_bf16(
                    false, fa_hi.v, false, fb_hi.v, (short)0, acc[t], false, false);
                acc[t] = __builtin_amdgcn_wmma_f32_16x16x32_bf16(
                    false, fa_hi.v, false, fb_lo.v, (short)0, acc[t], false, false);
                acc[t] = __builtin_amdgcn_wmma_f32_16x16x32_bf16(
                    false, fa_lo.v, false, fb_hi.v, (short)0, acc[t], false, false);
            }
            __syncthreads();  // done reading buf before it is refilled
        }

        // ---- fold distances into running per-lane min ----
        // C layout: VGPR r -> row (r + halfsel*8), lane ln -> column.
        #pragma unroll
        for (int t = 0; t < 8; ++t) {
            int col = ct + t * 16 + ln;
            float c2v = c2[col];
            #pragma unroll
            for (int r = 0; r < 8; ++r) {
                float d = __builtin_fmaf(-2.f, acc[t][r], c2v);
                if (d < minv[r]) { minv[r] = d; mini[r] = col; }
            }
        }
    }

    // ---- reduce across the 16 lanes sharing each row ----
    #pragma unroll
    for (int r = 0; r < 8; ++r) {
        #pragma unroll
        for (int m = 1; m < 16; m <<= 1) {
            float ov = __shfl_xor(minv[r], m, 32);
            int   oi = __shfl_xor(mini[r], m, 32);
            if (ov < minv[r] || (ov == minv[r] && oi < mini[r])) {
                minv[r] = ov; mini[r] = oi;
            }
        }
    }
    if (ln == 0) {
        #pragma unroll
        for (int r = 0; r < 8; ++r)
            outidx[rowbase + wave * 16 + halfsel * 8 + r] = mini[r];
    }
}

// ============================================================
// Kernel 3: gather z_q = codebook[idx], float indices, loss sum.
// ============================================================
__global__ void vq_gather(const float* __restrict__ h,
                          const float* __restrict__ cb,
                          const int* __restrict__ idx,
                          float* __restrict__ zq,
                          float* __restrict__ oidx,
                          float* __restrict__ wloss) {
    const int wave = threadIdx.x >> 5;
    const int lane = threadIdx.x & 31;
    const int n = blockIdx.x * 8 + wave;
    const int k = idx[n];
    const float* src = cb + (size_t)k * D_;
    const float* hr  = h  + (size_t)n * D_;
    float* dst = zq + (size_t)n * D_;
    float s = 0.f;
    #pragma unroll
    for (int j = 0; j < 16; ++j) {
        int d = lane + j * 32;
        float z = src[d];
        dst[d] = z;
        float df = hr[d] - z;
        s = __builtin_fmaf(df, df, s);
    }
    #pragma unroll
    for (int m = 16; m >= 1; m >>= 1) s += __shfl_xor(s, m, 32);
    if (lane == 0) {
        oidx[n] = (float)k;
        atomicAdd(wloss, s);
    }
}

// ============================================================
// Kernel 4: losses = sum / (N*D); commit == codebook in fwd.
// ============================================================
__global__ void vq_finalize(const float* __restrict__ wloss,
                            float* __restrict__ oloss) {
    float l = *wloss * (1.0f / ((float)N_ * (float)D_));
    oloss[0] = l;
    oloss[1] = l;
}

extern "C" void kernel_launch(void* const* d_in, const int* in_sizes, int n_in,
                              void* d_out, int out_size, void* d_ws, size_t ws_size,
                              hipStream_t stream) {
    (void)in_sizes; (void)n_in; (void)out_size; (void)ws_size;
    const float* h  = (const float*)d_in[0];   // [B,T,D] f32
    const float* cb = (const float*)d_in[1];   // [K,D]   f32

    float* out   = (float*)d_out;
    float* zq    = out;                         // N*D floats
    float* oidx  = out + (size_t)N_ * D_;       // N floats (indices as f32)
    float* oloss = oidx + N_;                   // 2 floats

    unsigned short* h_hi  = (unsigned short*)d_ws;               // N*D
    unsigned short* h_lo  = h_hi + (size_t)N_ * D_;              // N*D
    unsigned short* cb_hi = h_lo + (size_t)N_ * D_;              // K*D
    unsigned short* cb_lo = cb_hi + (size_t)K_ * D_;             // K*D
    float* c2    = (float*)(cb_lo + (size_t)K_ * D_);            // K
    int*   widx  = (int*)(c2 + K_);                              // N
    float* wloss = (float*)(widx + N_);                          // 1

    vq_split  <<<(N_ * (size_t)D_) / 1024, 256, 0, stream>>>(h,  h_hi,  h_lo);
    vq_split  <<<(K_ * (size_t)D_) / 1024, 256, 0, stream>>>(cb, cb_hi, cb_lo);
    vq_prep   <<<K_ / 8,   256, 0, stream>>>(cb, c2, wloss);
    vq_argmin <<<N_ / 128, 256, 0, stream>>>(h_hi, h_lo, cb_hi, cb_lo, c2, widx);
    vq_gather <<<N_ / 8,   256, 0, stream>>>(h, cb, widx, zq, oidx, wloss);
    vq_finalize<<<1, 1, 0, stream>>>(wloss, oloss);
}
